// VisionMamba_7017976562394
// MI455X (gfx1250) — matrix operand order repaired and checked by reading, not compiled, not run on hardware
//
#include <hip/hip_runtime.h>
#include <hip/hip_bf16.h>
#include <math.h>

typedef __attribute__((ext_vector_type(16))) _Float16 v16h;
typedef __attribute__((ext_vector_type(8)))  _Float16 v8h;
typedef __attribute__((ext_vector_type(8)))  float    v8f;

// ---------------- problem constants ----------------
constexpr int B_   = 2;
constexpr int NF_  = 4, PF_ = 4;
constexpr int PD_  = 8, PH_ = 16, PW_ = 16;
constexpr int ID_  = 32, IH_ = 128, IW_ = 128;
constexpr int GD_  = 4, GH_ = 8, GW_ = 8;
constexpr int L_   = NF_ * GD_ * GH_ * GW_;        // 1024
constexpr int PDIM_= PD_ * PH_ * PW_ * PF_;        // 8192
constexpr int EMB_ = 384;
constexpr int NL_  = 4;
constexpr int DI_  = 768;                          // 2*EMBED
constexpr int XZ_  = 1536;                         // 2*DI
constexpr int DS_  = 16, DTR_ = 24, DCONV_ = 4;
constexpr int LM_  = 768;
constexpr int ROWS_= B_ * LM_;                     // 1536
constexpr float EPS_ = 1e-5f;

// ---------------- helpers ----------------
__device__ __forceinline__ float block_reduce(float v, float* sm, int nt) {
  int tid = threadIdx.x;
  sm[tid] = v;
  __syncthreads();
  for (int s = nt >> 1; s > 0; s >>= 1) {
    if (tid < s) sm[tid] += sm[tid + s];
    __syncthreads();
  }
  float r = sm[0];
  __syncthreads();
  return r;
}

// ---------------- f32 -> f16 convert with zero pad ----------------
__global__ void k_convert_pad(const float* __restrict__ src, _Float16* __restrict__ dst,
                              int srows, int scols, int drows, int dcols) {
  int idx = blockIdx.x * blockDim.x + threadIdx.x;
  int total = drows * dcols;
  if (idx >= total) return;
  int r = idx / dcols, c = idx - r * dcols;
  float v = (r < srows && c < scols) ? src[(size_t)r * scols + c] : 0.0f;
  dst[idx] = (_Float16)v;
}

// ---------------- patchify + LayerNorm(PATCH_DIM) -> f16 ----------------
__global__ void k_patchify_ln1(const float* __restrict__ x,
                               const float* __restrict__ w, const float* __restrict__ bia,
                               _Float16* __restrict__ out) {
  int row = blockIdx.x;                 // b*L + l
  int b = row / L_, l = row - b * L_;
  int z  = l / (GD_ * GH_ * GW_);
  int rem = l - z * (GD_ * GH_ * GW_);
  int gd = rem / (GH_ * GW_); rem -= gd * (GH_ * GW_);
  int gh = rem / GW_;
  int gw = rem - gh * GW_;

  __shared__ float sm[256];
  float vals[PDIM_ / 256];
  float s = 0.f;
  #pragma unroll
  for (int i = 0; i < PDIM_ / 256; i++) {
    int p  = threadIdx.x + i * 256;     // p = ((pd*PH+ph)*PW+pw)*PF+pf  (C==1)
    int pf = p & 3;
    int t1 = p >> 2;
    int pw = t1 & 15;
    int t2 = t1 >> 4;
    int ph = t2 & 15;
    int pd = t2 >> 4;
    int tt = z * PF_ + pf;
    int dd = gd * PD_ + pd;
    int hh = gh * PH_ + ph;
    int ww = gw * PW_ + pw;
    size_t xi = ((((size_t)b * (NF_ * PF_) + tt) * ID_ + dd) * IH_ + hh) * IW_ + ww;
    float v = x[xi];
    vals[i] = v;
    s += v;
  }
  float mean = block_reduce(s, sm, 256) * (1.0f / PDIM_);
  float ss = 0.f;
  #pragma unroll
  for (int i = 0; i < PDIM_ / 256; i++) { float d = vals[i] - mean; ss += d * d; }
  float var = block_reduce(ss, sm, 256) * (1.0f / PDIM_);
  float rstd = rsqrtf(var + EPS_);
  #pragma unroll
  for (int i = 0; i < PDIM_ / 256; i++) {
    int p = threadIdx.x + i * 256;
    float o = (vals[i] - mean) * rstd * w[p] + bia[p];
    out[(size_t)row * PDIM_ + p] = (_Float16)o;
  }
}

// ------- WMMA NT GEMM: C[M,N] = A[M,K] * B[N,K]^T (+bias)(+softplus) -----------------
// One wave computes a 16x64 strip (4 accumulators): A fragment loaded once per k-step
// and reused across 4 WMMAs -> 10 b128 loads per 4 WMMAs instead of 16.
__global__ void __launch_bounds__(256)
k_gemm_nt_wmma(const _Float16* __restrict__ A, int lda,
               const _Float16* __restrict__ Bm, int ldb,
               const float* __restrict__ bias,
               float* __restrict__ C, int ldc,
               int M, int N, int K, int act) {
  int wave = threadIdx.x >> 5;
  int lane = threadIdx.x & 31;
  int tiles_n = N >> 6;                               // 64-wide strips
  int tile = blockIdx.x * (blockDim.x >> 5) + wave;
  if (tile >= (M >> 4) * tiles_n) return;             // wave-uniform: EXEC stays all ones
  int tm = tile / tiles_n, tn = tile - tm * tiles_n;
  int r  = lane & 15, hf = lane >> 4;

  // A fragment (16-bit 16x32): lanes<16 hold K 0..15 ; lanes>=16 hold 16..31
  const _Float16* ap = A  + (size_t)(tm * 16 + r) * lda + hf * 16;
  // B fragment: lane n holds 16 contiguous K (0..15 for lanes<16, 16..31 for lanes>=16)
  const _Float16* bp = Bm + (size_t)(tn * 64 + r) * ldb + hf * 16;
  size_t bstep = (size_t)16 * ldb;

  v8f acc[4];
  #pragma unroll
  for (int g = 0; g < 4; g++) acc[g] = (v8f){0.f,0.f,0.f,0.f,0.f,0.f,0.f,0.f};

  for (int k = 0; k < K; k += 32) {
    v8h a0 = *(const v8h*)(ap + k);
    v8h a1 = *(const v8h*)(ap + k + 8);
    v16h av = __builtin_shufflevector(a0, a1, 0,1,2,3,4,5,6,7,8,9,10,11,12,13,14,15);
    #pragma unroll
    for (int g = 0; g < 4; g++) {
      v16h bv = *(const v16h*)(bp + (size_t)g * bstep + k);
      acc[g] = __builtin_amdgcn_wmma_f32_16x16x32_f16(false, av, false, bv,
                                                      (short)0, acc[g], false, false);
    }
  }
  #pragma unroll
  for (int g = 0; g < 4; g++) {
    int col = tn * 64 + g * 16 + r;
    float badd = bias ? bias[col] : 0.0f;
    #pragma unroll
    for (int i = 0; i < 8; i++) {
      int rowi = tm * 16 + hf * 8 + i;
      float v = acc[g][i] + badd;
      if (act == 1) v = (v > 20.f) ? v : logf(1.f + __expf(v));   // softplus
      C[(size_t)rowi * ldc + col] = v;
    }
  }
}

// ---------------- LN2 over EMBED + gather(mask) + posemb; init residual=0 ------------
__global__ void k_ln2_gather(const float* __restrict__ e,
                             const float* __restrict__ w, const float* __restrict__ bia,
                             const int* __restrict__ mask, const float* __restrict__ pos,
                             float* __restrict__ hidden, float* __restrict__ residual) {
  int row = blockIdx.x;                 // b*LM + i
  int b = row / LM_, i = row - b * LM_;
  int jm = mask[i];
  const float* er = e + ((size_t)b * L_ + jm) * EMB_;
  __shared__ float sm[128];
  float vals[EMB_ / 128];
  float s = 0.f;
  #pragma unroll
  for (int t = 0; t < EMB_ / 128; t++) { vals[t] = er[threadIdx.x + t * 128]; s += vals[t]; }
  float mean = block_reduce(s, sm, 128) * (1.0f / EMB_);
  float ss = 0.f;
  #pragma unroll
  for (int t = 0; t < EMB_ / 128; t++) { float d = vals[t] - mean; ss += d * d; }
  float rstd = rsqrtf(block_reduce(ss, sm, 128) * (1.0f / EMB_) + EPS_);
  #pragma unroll
  for (int t = 0; t < EMB_ / 128; t++) {
    int c = threadIdx.x + t * 128;
    float o = (vals[t] - mean) * rstd * w[c] + bia[c] + pos[(size_t)jm * EMB_ + c];
    hidden[(size_t)row * EMB_ + c] = o;
    residual[(size_t)row * EMB_ + c] = 0.0f;
  }
}

// ------------- residual += hidden ; rmsnorm -> f16 (layers) or f32 (final) -----------
__global__ void k_add_rmsnorm(float* __restrict__ residual, const float* __restrict__ hidden,
                              const float* __restrict__ w,
                              _Float16* __restrict__ out16, float* __restrict__ out32) {
  int row = blockIdx.x;
  __shared__ float sm[128];
  float vals[EMB_ / 128];
  float ss = 0.f;
  #pragma unroll
  for (int t = 0; t < EMB_ / 128; t++) {
    int c = threadIdx.x + t * 128;
    float r = residual[(size_t)row * EMB_ + c] + hidden[(size_t)row * EMB_ + c];
    vals[t] = r; ss += r * r;
  }
  float rms = rsqrtf(block_reduce(ss, sm, 128) * (1.0f / EMB_) + EPS_);
  #pragma unroll
  for (int t = 0; t < EMB_ / 128; t++) {
    int c = threadIdx.x + t * 128;
    residual[(size_t)row * EMB_ + c] = vals[t];
    float o = vals[t] * rms * w[c];
    if (out16) out16[(size_t)row * EMB_ + c] = (_Float16)o;
    else       out32[(size_t)row * EMB_ + c] = o;
  }
}

// --------- depthwise causal conv (DCONV=4) over xi (fwd or time-reversed) + SiLU -----
__global__ void k_conv_silu(const float* __restrict__ xz,
                            const float* __restrict__ w, const float* __restrict__ bia,
                            float* __restrict__ uc, _Float16* __restrict__ uc16, int rev) {
  int idx = blockIdx.x * blockDim.x + threadIdx.x;
  if (idx >= ROWS_ * DI_) return;
  int d = idx % DI_;
  int t = (idx / DI_) % LM_;
  int b = idx / (DI_ * LM_);
  float acc = bia[d];
  #pragma unroll
  for (int j = 0; j < DCONV_; j++) {
    int tj = t - (DCONV_ - 1) + j;
    if (tj < 0) continue;
    int ts = rev ? (LM_ - 1 - tj) : tj;
    acc += w[d * DCONV_ + j] * xz[((size_t)b * LM_ + ts) * XZ_ + d];   // xi = xz[:, :DI]
  }
  float sv = acc / (1.0f + __expf(-acc));  // SiLU
  uc[idx]   = sv;
  uc16[idx] = (_Float16)sv;
}

// --------- dbl f32 (ldc=64) -> f16 [:,0:24] zero-padded to 32 for dt-GEMM ------------
__global__ void k_dbl_to_f16(const float* __restrict__ dbl, _Float16* __restrict__ out) {
  int idx = blockIdx.x * blockDim.x + threadIdx.x;
  if (idx >= ROWS_ * 32) return;
  int r = idx >> 5, c = idx & 31;
  out[idx] = (_Float16)(c < DTR_ ? dbl[(size_t)r * 64 + c] : 0.0f);
}

// --------- selective scan: one block per batch, 768 threads = one channel each -------
__global__ void __launch_bounds__(DI_) k_scan(const float* __restrict__ uc,
                                              const float* __restrict__ dt,
                                              const float* __restrict__ dbl,   // ldc=64, B at 24, C at 40
                                              const float* __restrict__ Alog,
                                              const float* __restrict__ Dp,
                                              float* __restrict__ y) {
  int b = blockIdx.x;
  int d = threadIdx.x;
  float A[DS_], h[DS_];
  #pragma unroll
  for (int s = 0; s < DS_; s++) { A[s] = -__expf(Alog[d * DS_ + s]); h[s] = 0.f; }
  float Dd = Dp[d];
  for (int t = 0; t < LM_; t++) {
    size_t row = (size_t)b * LM_ + t;
    float dtv = dt[row * DI_ + d];
    float uv  = uc[row * DI_ + d];
    float du  = dtv * uv;
    const float* bc = dbl + row * 64 + DTR_;
    float yv = 0.f;
    #pragma unroll
    for (int s = 0; s < DS_; s++) {
      float Bt = bc[s];
      float Ct = bc[DS_ + s];
      h[s] = h[s] * __expf(dtv * A[s]) + du * Bt;
      yv  += h[s] * Ct;
    }
    y[row * DI_ + d] = yv + uv * Dd;
  }
}

// --------- y = (yf + reverse(yb)) * silu(z) -> f16 for out_proj ----------------------
__global__ void k_combine_silu(const float* __restrict__ xz,
                               const float* __restrict__ yf, const float* __restrict__ yb,
                               _Float16* __restrict__ out) {
  int idx = blockIdx.x * blockDim.x + threadIdx.x;
  if (idx >= ROWS_ * DI_) return;
  int d = idx % DI_;
  int t = (idx / DI_) % LM_;
  int b = idx / (DI_ * LM_);
  size_t row  = (size_t)b * LM_ + t;
  size_t rrow = (size_t)b * LM_ + (LM_ - 1 - t);
  float z  = xz[row * XZ_ + DI_ + d];
  float sv = z / (1.0f + __expf(-z));
  float v  = (yf[row * DI_ + d] + yb[rrow * DI_ + d]) * sv;
  out[row * DI_ + d] = (_Float16)v;
}

// =====================================================================================
extern "C" void kernel_launch(void* const* d_in, const int* in_sizes, int n_in,
                              void* d_out, int out_size, void* d_ws, size_t ws_size,
                              hipStream_t stream) {
  // ---- inputs (setup_inputs dict order; layers dict flattened in key order) ----
  const float* x           = (const float*)d_in[0];
  const int*   mask        = (const int*)  d_in[1];
  const float* posemb      = (const float*)d_in[2];
  const float* ln1_w       = (const float*)d_in[3];
  const float* ln1_b       = (const float*)d_in[4];
  const float* plin_w      = (const float*)d_in[5];
  const float* plin_b      = (const float*)d_in[6];
  const float* ln2_w       = (const float*)d_in[7];
  const float* ln2_b       = (const float*)d_in[8];
  const float* in_proj_w   = (const float*)d_in[9];
  const float* conv_w_f    = (const float*)d_in[10];
  const float* conv_b_f    = (const float*)d_in[11];
  const float* xproj_w_f   = (const float*)d_in[12];
  const float* dtproj_w_f  = (const float*)d_in[13];
  const float* dtproj_b_f  = (const float*)d_in[14];
  const float* A_log_f     = (const float*)d_in[15];
  const float* D_f         = (const float*)d_in[16];
  const float* conv_w_b    = (const float*)d_in[17];
  const float* conv_b_b    = (const float*)d_in[18];
  const float* xproj_w_b   = (const float*)d_in[19];
  const float* dtproj_w_b  = (const float*)d_in[20];
  const float* dtproj_b_b  = (const float*)d_in[21];
  const float* A_log_b     = (const float*)d_in[22];
  const float* D_b         = (const float*)d_in[23];
  const float* out_proj_w  = (const float*)d_in[24];
  const float* norm_w      = (const float*)d_in[25];
  const float* norm_f_w    = (const float*)d_in[26];
  float* out = (float*)d_out;

  // ---- workspace carve-up (256B aligned) ----
  char* base = (char*)d_ws;
  size_t off = 0;
  auto carve = [&](size_t bytes) -> char* {
    char* p = base + off;
    off = (off + bytes + 255) & ~(size_t)255;
    return p;
  };
  _Float16* p16      = (_Float16*)carve((size_t)B_ * L_ * PDIM_ * 2);
  _Float16* wPatch16 = (_Float16*)carve((size_t)EMB_ * PDIM_ * 2);
  _Float16* wIn16    = (_Float16*)carve((size_t)NL_ * XZ_ * EMB_ * 2);
  _Float16* wOut16   = (_Float16*)carve((size_t)NL_ * EMB_ * DI_ * 2);
  _Float16* wX16[2]  = {(_Float16*)carve((size_t)NL_ * 64 * DI_ * 2),
                        (_Float16*)carve((size_t)NL_ * 64 * DI_ * 2)};
  _Float16* wDt16[2] = {(_Float16*)carve((size_t)NL_ * DI_ * 32 * 2),
                        (_Float16*)carve((size_t)NL_ * DI_ * 32 * 2)};
  float* eBuf     = (float*)carve((size_t)B_ * L_ * EMB_ * 4);
  float* hidden   = (float*)carve((size_t)ROWS_ * EMB_ * 4);
  float* residual = (float*)carve((size_t)ROWS_ * EMB_ * 4);
  _Float16* hn16  = (_Float16*)carve((size_t)ROWS_ * EMB_ * 2);
  float* xz       = (float*)carve((size_t)ROWS_ * XZ_ * 4);
  float* ucf[2]   = {(float*)carve((size_t)ROWS_ * DI_ * 4), (float*)carve((size_t)ROWS_ * DI_ * 4)};
  _Float16* uc16[2] = {(_Float16*)carve((size_t)ROWS_ * DI_ * 2), (_Float16*)carve((size_t)ROWS_ * DI_ * 2)};
  float* dblf[2]  = {(float*)carve((size_t)ROWS_ * 64 * 4), (float*)carve((size_t)ROWS_ * 64 * 4)};
  _Float16* dbl16[2] = {(_Float16*)carve((size_t)ROWS_ * 32 * 2), (_Float16*)carve((size_t)ROWS_ * 32 * 2)};
  float* dtb[2]   = {(float*)carve((size_t)ROWS_ * DI_ * 4), (float*)carve((size_t)ROWS_ * DI_ * 4)};
  float* ybuf[2]  = {(float*)carve((size_t)ROWS_ * DI_ * 4), (float*)carve((size_t)ROWS_ * DI_ * 4)};
  _Float16* yc16  = (_Float16*)carve((size_t)ROWS_ * DI_ * 2);
  (void)ws_size; (void)in_sizes; (void)n_in; (void)out_size;

  auto cvt = [&](const float* src, _Float16* dst, int sr, int sc, int dr, int dc) {
    int total = dr * dc;
    k_convert_pad<<<(total + 255) / 256, 256, 0, stream>>>(src, dst, sr, sc, dr, dc);
  };
  auto gemm = [&](const _Float16* A, int lda, const _Float16* Bw, int ldb,
                  const float* bias, float* C, int ldc, int M, int N, int K, int act) {
    int tiles = (M / 16) * (N / 64);     // one wave per 16x64 strip
    k_gemm_nt_wmma<<<(tiles + 7) / 8, 256, 0, stream>>>(A, lda, Bw, ldb, bias, C, ldc, M, N, K, act);
  };

  // ---- weight conversion ----
  cvt(plin_w,     wPatch16, EMB_, PDIM_, EMB_, PDIM_);
  cvt(in_proj_w,  wIn16,    NL_ * XZ_, EMB_, NL_ * XZ_, EMB_);
  cvt(out_proj_w, wOut16,   NL_ * EMB_, DI_, NL_ * EMB_, DI_);
  for (int i = 0; i < NL_; i++) {
    cvt(xproj_w_f + (size_t)i * 56 * DI_, wX16[0] + (size_t)i * 64 * DI_, 56, DI_, 64, DI_);
    cvt(xproj_w_b + (size_t)i * 56 * DI_, wX16[1] + (size_t)i * 64 * DI_, 56, DI_, 64, DI_);
  }
  cvt(dtproj_w_f, wDt16[0], NL_ * DI_, DTR_, NL_ * DI_, 32);
  cvt(dtproj_w_b, wDt16[1], NL_ * DI_, DTR_, NL_ * DI_, 32);

  // ---- patchify + LN1 -> f16 ----
  k_patchify_ln1<<<B_ * L_, 256, 0, stream>>>(x, ln1_w, ln1_b, p16);

  // ---- big patch GEMM: (2048 x 384 x 8192) + bias ----
  gemm(p16, PDIM_, wPatch16, PDIM_, plin_b, eBuf, EMB_, B_ * L_, EMB_, PDIM_, 0);

  // ---- LN2 + gather + posemb ----
  k_ln2_gather<<<ROWS_, 128, 0, stream>>>(eBuf, ln2_w, ln2_b, mask, posemb, hidden, residual);

  const float* cw[2]  = {conv_w_f, conv_w_b};
  const float* cb[2]  = {conv_b_f, conv_b_b};
  const float* dtbia[2] = {dtproj_b_f, dtproj_b_b};
  const float* alog[2]  = {A_log_f, A_log_b};
  const float* dvec[2]  = {D_f, D_b};

  for (int i = 0; i < NL_; i++) {
    // residual += hidden ; h = rmsnorm(residual) -> f16
    k_add_rmsnorm<<<ROWS_, 128, 0, stream>>>(residual, hidden, norm_w + (size_t)i * EMB_,
                                             hn16, nullptr);
    // in_proj: (1536 x 1536 x 384)
    gemm(hn16, EMB_, wIn16 + (size_t)i * XZ_ * EMB_, EMB_, nullptr, xz, XZ_,
         ROWS_, XZ_, EMB_, 0);

    for (int br = 0; br < 2; br++) {
      int total = ROWS_ * DI_;
      k_conv_silu<<<(total + 255) / 256, 256, 0, stream>>>(
          xz, cw[br] + (size_t)i * DI_ * DCONV_, cb[br] + (size_t)i * DI_,
          ucf[br], uc16[br], br);
      // xproj: (1536 x 64 x 768), first 56 cols meaningful
      gemm(uc16[br], DI_, wX16[br] + (size_t)i * 64 * DI_, DI_, nullptr,
           dblf[br], 64, ROWS_, 64, DI_, 0);
      k_dbl_to_f16<<<(ROWS_ * 32 + 255) / 256, 256, 0, stream>>>(dblf[br], dbl16[br]);
      // dt: (1536 x 768 x 32) + bias, softplus
      gemm(dbl16[br], 32, wDt16[br] + (size_t)i * DI_ * 32, 32,
           dtbia[br] + (size_t)i * DI_, dtb[br], DI_, ROWS_, DI_, 32, 1);
      // selective scan
      k_scan<<<B_, DI_, 0, stream>>>(ucf[br], dtb[br], dblf[br],
                                     alog[br] + (size_t)i * DI_ * DS_,
                                     dvec[br] + (size_t)i * DI_, ybuf[br]);
    }
    // (yf + rev(yb)) * silu(z) -> f16
    k_combine_silu<<<(ROWS_ * DI_ + 255) / 256, 256, 0, stream>>>(xz, ybuf[0], ybuf[1], yc16);
    // out_proj: (1536 x 384 x 768) -> hidden
    gemm(yc16, DI_, wOut16 + (size_t)i * EMB_ * DI_, DI_, nullptr, hidden, EMB_,
         ROWS_, EMB_, DI_, 0);
  }

  // final: residual += hidden ; rmsnorm -> d_out (f32)
  k_add_rmsnorm<<<ROWS_, 128, 0, stream>>>(residual, hidden, norm_f_w, nullptr, out);
}
